// AttentionBlock_11811160064661
// MI455X (gfx1250) — matrix operand order repaired and checked by reading, not compiled, not run on hardware
//
#include <hip/hip_runtime.h>

// ---------------------------------------------------------------------------
// CDNA5 (gfx1250) AttentionBlock: GroupNorm -> QKV -> MHA -> OutProj+residual
// Matmuls: v_wmma_f32_16x16x32_bf16 (wave32).
// GEMM tile staging: Tensor Data Mover (tensor_load_to_lds) + s_wait_tensorcnt,
// double-buffered so TDM copy of tile k+1 overlaps WMMA on tile k.
// ---------------------------------------------------------------------------

typedef __attribute__((ext_vector_type(16))) __bf16 v16bf;
typedef __attribute__((ext_vector_type(8)))  float  v8f;
typedef __attribute__((ext_vector_type(4)))  unsigned int u32x4;

typedef unsigned int u32x4v __attribute__((ext_vector_type(4)));
typedef int          i32x4v __attribute__((ext_vector_type(4)));
typedef int          i32x8v __attribute__((ext_vector_type(8)));

union V16BF { v16bf v; u32x4 q[2]; };

// A-fragment (16x32 bf16): per-lane row, two contiguous 8-elem k runs.
__device__ __forceinline__ v16bf load_frag_2x16B(const __bf16* p0, const __bf16* p1) {
  V16BF t;
  t.q[0] = *reinterpret_cast<const u32x4*>(p0);
  t.q[1] = *reinterpret_cast<const u32x4*>(p1);
  return t.v;
}
// B-fragment (32x16 bf16): per-lane column, 16 contiguous k values.
__device__ __forceinline__ v16bf load_frag_32B(const __bf16* p) {
  V16BF t;
  t.q[0] = *reinterpret_cast<const u32x4*>(p);
  t.q[1] = *reinterpret_cast<const u32x4*>(p + 8);
  return t.v;
}

__device__ __forceinline__ v8f wmma_bf16(v16bf a, v16bf b, v8f c) {
  return __builtin_amdgcn_wmma_f32_16x16x32_bf16(false, a, false, b,
                                                 (short)0, c, false, false);
}

// ---------------------------------------------------------------------------
// TDM: issue a 2D tile load (tile1 rows x tile0 elems, 2-byte elements) from a
// row-major tensor (row stride = stride0 elems) into LDS at lds_off.
// D# layout per CDNA5 ISA.
// ---------------------------------------------------------------------------
__device__ __forceinline__ void tdm_load_2d_bf16(unsigned lds_off, const void* gaddr,
                                                 unsigned dim0, unsigned dim1,
                                                 unsigned stride0,
                                                 unsigned tile0, unsigned tile1) {
  const unsigned long long ga = (unsigned long long)gaddr;
  u32x4v g0;
  g0.x = 1u;                                            // count=1, user mode
  g0.y = lds_off;                                       // lds_addr [63:32]
  g0.z = (unsigned)(ga & 0xffffffffu);                  // global_addr [95:64]
  g0.w = (unsigned)((ga >> 32) & 0x01ffffffu) | 0x80000000u;  // addr hi | type=2

  i32x8v g1;
  g1[0] = 0x00010000;                                   // data_size=1 (2 bytes)
  g1[1] = (int)((dim0 & 0xffffu) << 16);                // tensor_dim0 lo (bit 48)
  g1[2] = (int)((dim0 >> 16) | ((dim1 & 0xffffu) << 16));  // dim0 hi | dim1 lo
  g1[3] = (int)((dim1 >> 16) | (tile0 << 16));          // dim1 hi | tile_dim0
  g1[4] = (int)(tile1 & 0xffffu);                       // tile_dim1 | tile_dim2=0
  g1[5] = (int)stride0;                                 // tensor_dim0_stride lo
  g1[6] = 0;                                            // stride0 hi | stride1 lo
  g1[7] = 0;

  i32x4v z4 = {};
#if __has_include(<hip/amd_detail/amd_gfx1250_TDM.h>)
  i32x8v z8 = {};
  __builtin_amdgcn_tensor_load_to_lds(g0, g1, z4, z4, z8, 0);
#else
  __builtin_amdgcn_tensor_load_to_lds(g0, g1, z4, z4, 0);
#endif
}

__device__ __forceinline__ unsigned lds_offset_of(const void* shared_ptr) {
  // LDS aperture: generic address low 32 bits are the LDS byte offset.
  return (unsigned)(unsigned long long)shared_ptr;
}

#define BATCH 8
#define CHAN  512
#define SPAT  1024
#define NHEAD 8
#define HDIM  64

// ---------------------------------------------------------------------------
// Kernel 1: GroupNorm, write bf16 activations in [B*S, C] layout (GEMM-ready)
// ---------------------------------------------------------------------------
__global__ __launch_bounds__(256) void gn_kernel(const float* __restrict__ x,
                                                 const float* __restrict__ gamma,
                                                 const float* __restrict__ beta,
                                                 __bf16* __restrict__ h_bf) {
  const int bg  = blockIdx.x;        // B*32 blocks
  const int b   = bg >> 5;
  const int g   = bg & 31;
  const int tid = threadIdx.x;
  const float* xg = x + ((size_t)b * CHAN + g * 16) * SPAT;   // 16 ch x 1024

  float s1 = 0.f, s2 = 0.f;
  for (int i = tid; i < 16384; i += 256) {
    float v = xg[i];
    s1 += v; s2 += v * v;
  }
  __shared__ float r1[256], r2[256];
  r1[tid] = s1; r2[tid] = s2;
  __syncthreads();
  for (int off = 128; off > 0; off >>= 1) {
    if (tid < off) { r1[tid] += r1[tid + off]; r2[tid] += r2[tid + off]; }
    __syncthreads();
  }
  const float mean = r1[0] * (1.f / 16384.f);
  const float var  = r2[0] * (1.f / 16384.f) - mean * mean;
  const float rstd = rsqrtf(var + 1e-6f);

  for (int i = tid; i < 16384; i += 256) {
    const int cl = i & 15;            // channel-in-group fastest -> coalesced stores
    const int s  = i >> 4;
    const int c  = g * 16 + cl;
    const float v = (xg[(size_t)cl * SPAT + s] - mean) * rstd * gamma[c] + beta[c];
    h_bf[((size_t)b * SPAT + s) * CHAN + c] = (__bf16)v;
  }
}

// ---------------------------------------------------------------------------
// Kernel 2: f32 -> bf16 weight conversion
// ---------------------------------------------------------------------------
__global__ __launch_bounds__(256) void cvt_kernel(const float* __restrict__ src,
                                                  __bf16* __restrict__ dst, int n) {
  const int i = (blockIdx.x * 256 + threadIdx.x) * 4;
  if (i + 3 < n) {
#pragma unroll
    for (int j = 0; j < 4; ++j) dst[i + j] = (__bf16)src[i + j];
  }
}

// ---------------------------------------------------------------------------
// Kernel 3: QKV GEMM.  y = h @ W^T.  128x128 block tile, 8 waves of 64x32.
// A/B 128x32 tiles double-buffered in LDS, filled by the TDM.
//   z=0: Q (scaled 1/sqrt(hd)) -> [bh][S][hd]
//   z=1: K                      -> [bh][S][hd]
//   z=2: V transposed           -> [bh][hd][S]
// ---------------------------------------------------------------------------
__global__ __launch_bounds__(256) void qkv_kernel(const __bf16* __restrict__ h_bf,
                                                  const __bf16* __restrict__ wq_bf,
                                                  const __bf16* __restrict__ wk_bf,
                                                  const __bf16* __restrict__ wv_bf,
                                                  __bf16* __restrict__ qo,
                                                  __bf16* __restrict__ ko,
                                                  __bf16* __restrict__ vto) {
  __shared__ __bf16 At[2][128 * 32];
  __shared__ __bf16 Bt[2][128 * 32];

  const int tid  = threadIdx.x;
  const int lane = tid & 31;
  const int w    = tid >> 5;
  const int half = lane >> 4;
  const int l16  = lane & 15;
  const int koff = half * 8;
  const int mbase = blockIdx.x * 128;
  const int nbase = blockIdx.y * 128;
  const int mat   = blockIdx.z;
  const __bf16* wsrc = (mat == 0) ? wq_bf : (mat == 1) ? wk_bf : wv_bf;

  const int wm = (w >> 2) * 64;
  const int wn = (w & 3) * 32;
  const int NK = CHAN / 32;                       // 16 k-steps

  v8f zero = {};
  v8f acc[4][2];
#pragma unroll
  for (int i = 0; i < 4; ++i)
#pragma unroll
    for (int j = 0; j < 2; ++j) acc[i][j] = zero;

  // prologue: TDM fills buffer 0 with tile k=0
  if (w == 0) {
    tdm_load_2d_bf16(lds_offset_of(&At[0][0]), h_bf + (size_t)mbase * CHAN,
                     CHAN, BATCH * SPAT, CHAN, 32, 128);
    tdm_load_2d_bf16(lds_offset_of(&Bt[0][0]), wsrc + (size_t)nbase * CHAN,
                     CHAN, CHAN, CHAN, 32, 128);
  }

  for (int kb = 0; kb < NK; ++kb) {
    const int cur = kb & 1;
    if (w == 0) {
      if (kb + 1 < NK) {
        const int nxt = (kb + 1) & 1;
        const int kc  = (kb + 1) * 32;
        tdm_load_2d_bf16(lds_offset_of(&At[nxt][0]),
                         h_bf + (size_t)mbase * CHAN + kc,
                         CHAN, BATCH * SPAT, CHAN, 32, 128);
        tdm_load_2d_bf16(lds_offset_of(&Bt[nxt][0]),
                         wsrc + (size_t)nbase * CHAN + kc,
                         CHAN, CHAN, CHAN, 32, 128);
        __builtin_amdgcn_s_wait_tensorcnt((short)2);  // tile kb done (in-order)
      } else {
        __builtin_amdgcn_s_wait_tensorcnt((short)0);
      }
    }
    __syncthreads();                                  // tile kb visible to all

    v16bf a[4], bf[2];
#pragma unroll
    for (int i = 0; i < 4; ++i) {
      const __bf16* ap = &At[cur][0] + (wm + i * 16 + l16) * 32;
      a[i] = load_frag_2x16B(ap + half * 16, ap + half * 16 + 8);
    }
#pragma unroll
    for (int j = 0; j < 2; ++j)
      bf[j] = load_frag_32B(&Bt[cur][0] + (wn + j * 16 + l16) * 32 + half * 16);

#pragma unroll
    for (int i = 0; i < 4; ++i)
#pragma unroll
      for (int j = 0; j < 2; ++j)
        acc[i][j] = wmma_bf16(a[i], bf[j], acc[i][j]);

    __syncthreads();                 // all reads of buf[cur] done before reuse
  }

  const float qscale = (mat == 0) ? 0.125f : 1.0f;   // 1/sqrt(64)
#pragma unroll
  for (int i = 0; i < 4; ++i) {
#pragma unroll
    for (int j = 0; j < 2; ++j) {
#pragma unroll
      for (int e = 0; e < 8; ++e) {
        const int m = mbase + wm + i * 16 + e + half * 8;   // C-layout row
        const int n = nbase + wn + j * 16 + l16;            // C-layout col
        const int b = m >> 10, s = m & 1023;
        const int hh = n >> 6, d = n & 63;
        const __bf16 val = (__bf16)(acc[i][j][e] * qscale);
        if (mat == 2)
          vto[(((size_t)b * NHEAD + hh) * HDIM + d) * SPAT + s] = val;
        else if (mat == 0)
          qo[(((size_t)b * NHEAD + hh) * SPAT + s) * HDIM + d] = val;
        else
          ko[(((size_t)b * NHEAD + hh) * SPAT + s) * HDIM + d] = val;
      }
    }
  }
}

// ---------------------------------------------------------------------------
// Kernel 4: attention per (b,h).  8 waves; each wave flash-processes 16-row
// query strips: scores = Q K^T (WMMA), online softmax (half-wave shuffles),
// P reshaped via per-wave LDS tile into A-fragments, out += P V (WMMA).
// ---------------------------------------------------------------------------
__global__ __launch_bounds__(256) void attn_kernel(const __bf16* __restrict__ q,
                                                   const __bf16* __restrict__ kk,
                                                   const __bf16* __restrict__ vt,
                                                   __bf16* __restrict__ att) {
  __shared__ __bf16 P[8][16][64];                  // per-wave P tile, 16 KB

  const int bh   = blockIdx.x;                     // B*NHEAD = 64
  const int b    = bh >> 3, hh = bh & 7;
  const int tid  = threadIdx.x;
  const int lane = tid & 31;
  const int w    = tid >> 5;
  const int half = lane >> 4;
  const int l16  = lane & 15;
  const int koff = half * 8;

  const __bf16* Qb = q  + (size_t)bh * SPAT * HDIM;
  const __bf16* Kb = kk + (size_t)bh * SPAT * HDIM;
  const __bf16* Vb = vt + (size_t)bh * HDIM * SPAT;

  for (int strip = 0; strip < 8; ++strip) {
    const int qs = strip * 128 + w * 16;
    const __bf16* qrow = Qb + (size_t)(qs + l16) * HDIM;
    const v16bf a0 = load_frag_2x16B(qrow + half * 16,      qrow + half * 16 + 8);
    const v16bf a1 = load_frag_2x16B(qrow + 32 + half * 16, qrow + 40 + half * 16);

    v8f zero = {};
    v8f oacc[4];
#pragma unroll
    for (int j = 0; j < 4; ++j) oacc[j] = zero;
    float rmax[8], rsum[8];
#pragma unroll
    for (int e = 0; e < 8; ++e) { rmax[e] = -3.0e38f; rsum[e] = 0.f; }

    for (int kc = 0; kc < SPAT; kc += 64) {
      // ---- scores: 16x64 tile, hd=64 as two k=32 WMMA chunks ----
      v8f sc[4];
#pragma unroll
      for (int j = 0; j < 4; ++j) {
        const __bf16* krow = Kb + (size_t)(kc + j * 16 + l16) * HDIM;
        const v16bf b0 = load_frag_32B(krow + half * 16);
        const v16bf b1 = load_frag_32B(krow + 32 + half * 16);
        sc[j] = zero;
        sc[j] = wmma_bf16(a0, b0, sc[j]);
        sc[j] = wmma_bf16(a1, b1, sc[j]);
      }
      // ---- online softmax; row e+8*half lives across lanes (l16 = col) ----
#pragma unroll
      for (int e = 0; e < 8; ++e) {
        float mx = fmaxf(fmaxf(sc[0][e], sc[1][e]), fmaxf(sc[2][e], sc[3][e]));
#pragma unroll
        for (int msk = 8; msk >= 1; msk >>= 1) mx = fmaxf(mx, __shfl_xor(mx, msk, 32));
        const float nm    = fmaxf(rmax[e], mx);
        const float scale = __expf(rmax[e] - nm);
        rmax[e] = nm;
        float ps = 0.f;
#pragma unroll
        for (int j = 0; j < 4; ++j) {
          const float pv = __expf(sc[j][e] - nm);
          sc[j][e] = pv;
          ps += pv;
        }
#pragma unroll
        for (int msk = 8; msk >= 1; msk >>= 1) ps += __shfl_xor(ps, msk, 32);
        rsum[e] = rsum[e] * scale + ps;
#pragma unroll
        for (int j = 0; j < 4; ++j) oacc[j][e] *= scale;
        // C-layout -> LDS (row e+8*half, col j*16+l16)
#pragma unroll
        for (int j = 0; j < 4; ++j)
          P[w][e + half * 8][j * 16 + l16] = (__bf16)sc[j][e];
      }
      // ---- out += P @ V : reload P as A-fragments (same wave, DS in-order) --
      const __bf16* prow = &P[w][l16][0];
      const v16bf pa0 = load_frag_2x16B(prow + half * 16,      prow + half * 16 + 8);
      const v16bf pa1 = load_frag_2x16B(prow + 32 + half * 16, prow + 40 + half * 16);
#pragma unroll
      for (int j = 0; j < 4; ++j) {
        const __bf16* vrow = Vb + (size_t)(j * 16 + l16) * SPAT + kc;
        const v16bf vb0 = load_frag_32B(vrow + half * 16);
        const v16bf vb1 = load_frag_32B(vrow + 32 + half * 16);
        oacc[j] = wmma_bf16(pa0, vb0, oacc[j]);
        oacc[j] = wmma_bf16(pa1, vb1, oacc[j]);
      }
    }
    // ---- normalize + store [B*S, C] for the output projection ----
#pragma unroll
    for (int j = 0; j < 4; ++j)
#pragma unroll
      for (int e = 0; e < 8; ++e) {
        const int row = qs + e + half * 8;
        const int d   = j * 16 + l16;
        att[((size_t)b * SPAT + row) * CHAN + hh * HDIM + d] =
            (__bf16)(oacc[j][e] / rsum[e]);
      }
  }
}

// ---------------------------------------------------------------------------
// Kernel 5: output projection + bias + residual, fp32 out in [B,C,H,W].
// Same TDM double-buffered WMMA GEMM.
// ---------------------------------------------------------------------------
__global__ __launch_bounds__(256) void oproj_kernel(const __bf16* __restrict__ att,
                                                    const __bf16* __restrict__ wo_bf,
                                                    const float* __restrict__ bo,
                                                    const float* __restrict__ x,
                                                    float* __restrict__ out) {
  __shared__ __bf16 At[2][128 * 32];
  __shared__ __bf16 Bt[2][128 * 32];

  const int tid  = threadIdx.x;
  const int lane = tid & 31;
  const int w    = tid >> 5;
  const int half = lane >> 4;
  const int l16  = lane & 15;
  const int koff = half * 8;
  const int mbase = blockIdx.x * 128;
  const int nbase = blockIdx.y * 128;
  const int wm = (w >> 2) * 64;
  const int wn = (w & 3) * 32;
  const int NK = CHAN / 32;

  v8f zero = {};
  v8f acc[4][2];
#pragma unroll
  for (int i = 0; i < 4; ++i)
#pragma unroll
    for (int j = 0; j < 2; ++j) acc[i][j] = zero;

  if (w == 0) {
    tdm_load_2d_bf16(lds_offset_of(&At[0][0]), att + (size_t)mbase * CHAN,
                     CHAN, BATCH * SPAT, CHAN, 32, 128);
    tdm_load_2d_bf16(lds_offset_of(&Bt[0][0]), wo_bf + (size_t)nbase * CHAN,
                     CHAN, CHAN, CHAN, 32, 128);
  }

  for (int kb = 0; kb < NK; ++kb) {
    const int cur = kb & 1;
    if (w == 0) {
      if (kb + 1 < NK) {
        const int nxt = (kb + 1) & 1;
        const int kc  = (kb + 1) * 32;
        tdm_load_2d_bf16(lds_offset_of(&At[nxt][0]),
                         att + (size_t)mbase * CHAN + kc,
                         CHAN, BATCH * SPAT, CHAN, 32, 128);
        tdm_load_2d_bf16(lds_offset_of(&Bt[nxt][0]),
                         wo_bf + (size_t)nbase * CHAN + kc,
                         CHAN, CHAN, CHAN, 32, 128);
        __builtin_amdgcn_s_wait_tensorcnt((short)2);
      } else {
        __builtin_amdgcn_s_wait_tensorcnt((short)0);
      }
    }
    __syncthreads();

    v16bf a[4], bf[2];
#pragma unroll
    for (int i = 0; i < 4; ++i) {
      const __bf16* ap = &At[cur][0] + (wm + i * 16 + l16) * 32;
      a[i] = load_frag_2x16B(ap + half * 16, ap + half * 16 + 8);
    }
#pragma unroll
    for (int j = 0; j < 2; ++j)
      bf[j] = load_frag_32B(&Bt[cur][0] + (wn + j * 16 + l16) * 32 + half * 16);

#pragma unroll
    for (int i = 0; i < 4; ++i)
#pragma unroll
      for (int j = 0; j < 2; ++j)
        acc[i][j] = wmma_bf16(a[i], bf[j], acc[i][j]);

    __syncthreads();
  }

#pragma unroll
  for (int i = 0; i < 4; ++i)
#pragma unroll
    for (int j = 0; j < 2; ++j)
#pragma unroll
      for (int e = 0; e < 8; ++e) {
        const int m = mbase + wm + i * 16 + e + half * 8;
        const int n = nbase + wn + j * 16 + l16;
        const int b = m >> 10, s = m & 1023;
        const size_t idx = ((size_t)b * CHAN + n) * SPAT + s;   // [B,C,H*W]
        out[idx] = acc[i][j][e] + bo[n] + x[idx];
      }
}

// ---------------------------------------------------------------------------
// Launch
// ---------------------------------------------------------------------------
extern "C" void kernel_launch(void* const* d_in, const int* in_sizes, int n_in,
                              void* d_out, int out_size, void* d_ws, size_t ws_size,
                              hipStream_t stream) {
  const float* x     = (const float*)d_in[0];
  const float* wq    = (const float*)d_in[1];
  const float* wk    = (const float*)d_in[2];
  const float* wv    = (const float*)d_in[3];
  const float* wo    = (const float*)d_in[4];
  const float* bo    = (const float*)d_in[5];
  const float* gamma = (const float*)d_in[6];
  const float* beta  = (const float*)d_in[7];
  float* out = (float*)d_out;

  const size_t ACT  = (size_t)BATCH * SPAT * CHAN * sizeof(unsigned short);  // 8 MB
  const size_t WSZ  = (size_t)CHAN * CHAN * sizeof(unsigned short);          // 0.5 MB
  if (ws_size < 5 * ACT + 4 * WSZ + 4096) return;

  char* ws = (char*)d_ws;
  size_t off = 0;
  auto take = [&](size_t bytes) -> char* {
    char* p = ws + off;
    off += (bytes + 255) & ~(size_t)255;
    return p;
  };
  __bf16* h_bf  = (__bf16*)take(ACT);
  __bf16* qb    = (__bf16*)take(ACT);
  __bf16* kb    = (__bf16*)take(ACT);
  __bf16* vtb   = (__bf16*)take(ACT);
  __bf16* attb  = (__bf16*)take(ACT);
  __bf16* wq_bf = (__bf16*)take(WSZ);
  __bf16* wk_bf = (__bf16*)take(WSZ);
  __bf16* wv_bf = (__bf16*)take(WSZ);
  __bf16* wo_bf = (__bf16*)take(WSZ);

  // 1) GroupNorm -> bf16 [B*S, C]
  gn_kernel<<<BATCH * 32, 256, 0, stream>>>(x, gamma, beta, h_bf);

  // 2) weights -> bf16
  const int wn = CHAN * CHAN;                   // 262144
  const int wblk = wn / (256 * 4);              // 256 blocks
  cvt_kernel<<<wblk, 256, 0, stream>>>(wq, wq_bf, wn);
  cvt_kernel<<<wblk, 256, 0, stream>>>(wk, wk_bf, wn);
  cvt_kernel<<<wblk, 256, 0, stream>>>(wv, wv_bf, wn);
  cvt_kernel<<<wblk, 256, 0, stream>>>(wo, wo_bf, wn);

  // 3) QKV projections (z = matrix index)
  qkv_kernel<<<dim3(BATCH * SPAT / 128, CHAN / 128, 3), 256, 0, stream>>>(
      h_bf, wq_bf, wk_bf, wv_bf, qb, kb, vtb);

  // 4) attention per (b, head)
  attn_kernel<<<BATCH * NHEAD, 256, 0, stream>>>(qb, kb, vtb, attb);

  // 5) output projection + bias + residual
  oproj_kernel<<<dim3(BATCH * SPAT / 128, CHAN / 128), 256, 0, stream>>>(
      attb, wo_bf, bo, x, out);
}
